// Matformer4Stem_50508815401616
// MI455X (gfx1250) — compile-verified, matrix operand order checked by reading, not run on hardware
//
#include <hip/hip_runtime.h>
#include <stdint.h>
#include <stddef.h>

// ---------------------------------------------------------------------------
// Types / helpers for CDNA5 WMMA (gfx1250, wave32)
// ---------------------------------------------------------------------------
typedef __attribute__((ext_vector_type(16))) __bf16 v16bf;
typedef __attribute__((ext_vector_type(8)))  float  v8f;

union Frag16 { v16bf v; uint32_t u[8]; uint4 q[2]; };

__device__ __forceinline__ uint16_t f2bf(float f) {
  union { float f; uint32_t u; } x; x.f = f;
  uint32_t r = x.u + 0x7FFFu + ((x.u >> 16) & 1u);  // RNE
  return (uint16_t)(r >> 16);
}
__device__ __forceinline__ float bf2f(uint16_t b) {
  union { uint32_t u; float f; } x; x.u = ((uint32_t)b) << 16;
  return x.f;
}
// K-offset of fragment dword v for 16-bit A/B operands (ISA 7.12.2 layout):
// VGPR v holds K = 16*(v/4) + 2*(v%4) + 8*(lane/16), and K+1.
__device__ __host__ __forceinline__ int kb_of(int v, int half) {
  return ((v >> 2) << 4) + ((v & 3) << 1) + (half << 3);
}

// CDNA5 async global->LDS copy, 16 bytes per lane (ASYNCcnt-tracked).
__device__ __forceinline__ void async_copy_b128(uint32_t lds_off,
                                                const void* gptr) {
  asm volatile("global_load_async_to_lds_b128 %0, %1, off"
               :
               : "v"(lds_off), "v"(gptr)
               : "memory");
}
__device__ __forceinline__ void wait_async() {
  asm volatile("s_wait_asynccnt 0x0" ::: "memory");
}
__device__ __forceinline__ uint32_t lds_addr(const void* p) {
  return (uint32_t)(uintptr_t)p;  // LDS aperture: low 32 bits = LDS offset
}

// ---------------------------------------------------------------------------
// Weight packing, fragment-major: fp32 W[K][N] -> uint32[(K/32)*N*16].
// P[((kt*N + n)*16) + half*8 + v] = bf16(W[kb][n]) | bf16(W[kb+1][n])<<16
// with kb = kt*32 + kb_of(v, half). A lane's whole B fragment is then 8
// contiguous dwords -> two aligned global_load_b128 per (ntile, k-step).
// ---------------------------------------------------------------------------
__global__ void pack_weight_kernel(const float* __restrict__ W,
                                   uint32_t* __restrict__ P, int K, int N) {
  int idx = blockIdx.x * 256 + threadIdx.x;
  int total = (K >> 1) * N;
  if (idx >= total) return;
  int frag = idx & 15;  // half*8 + v
  int rest = idx >> 4;
  int n = rest % N;
  int kt = rest / N;
  int half = frag >> 3;
  int v = frag & 7;
  int kb = kt * 32 + kb_of(v, half);
  uint32_t lo = f2bf(W[(size_t)kb * N + n]);
  uint32_t hi = f2bf(W[(size_t)(kb + 1) * N + n]);
  P[idx] = lo | (hi << 16);
}

// Load one B fragment (two b128s) from fragment-major packed weights.
__device__ __forceinline__ void load_bfrag(Frag16& bf,
                                           const uint32_t* __restrict__ Bp,
                                           int N, int kt, int n, int half) {
  const uint4* p =
      (const uint4*)(Bp + ((size_t)kt * N + n) * 16 + (half << 3));
  bf.q[0] = p[0];
  bf.q[1] = p[1];
}

__global__ void f32_to_bf16_kernel(const float* __restrict__ in,
                                   uint16_t* __restrict__ out, int total) {
  int idx = blockIdx.x * 256 + threadIdx.x;
  if (idx < total) out[idx] = f2bf(in[idx]);
}

// ---------------------------------------------------------------------------
// Generic bf16 WMMA GEMM: out[M,N] = A[M,K] @ B[K,N] (+bias), MODE (template):
//   0: fp32 out + bias, 1: bf16 out + bias, 2: bf16 softplus + bias,
//   3: bf16 out, no bias
// Block = 8 waves covering mw m-tiles x nw n-waves (mw*nw == 8, nw*64 == N).
// A tile (mw*16 x K bf16) staged in LDS with async b128 copies.
// ---------------------------------------------------------------------------
template <int MODE>
__global__ __launch_bounds__(256) void gemm_bf16_kernel(
    const uint16_t* __restrict__ A, const uint32_t* __restrict__ Bp,
    const float* __restrict__ bias, float* __restrict__ outF,
    uint16_t* __restrict__ outB, int M, int N, int K, int mw, int nw) {
  extern __shared__ __align__(16) uint16_t sAt[];  // [mw*16][K]
  const int tid = threadIdx.x;
  const int wid = tid >> 5;
  const int lane = tid & 31;
  const int nl = lane & 15;
  const int half = lane >> 4;

  // ---- async-stage the A tile into LDS (coalesced 16B chunks) ----
  const int mrows = mw << 4;
  const int rowBase = blockIdx.x * mrows;
  const int chunksPerRow = K >> 3;  // 16B chunks of 8 bf16
  const int totalChunks = mrows * chunksPerRow;
  for (int ci = tid; ci < totalChunks; ci += 256) {
    int r = ci / chunksPerRow;
    int kc = ci - r * chunksPerRow;
    int rg = rowBase + r;
    if (rg >= M) rg = M - 1;  // clamp (stores are guarded below)
    const uint16_t* g = A + (size_t)rg * K + (kc << 3);
    async_copy_b128(lds_addr(sAt + (size_t)r * K + (kc << 3)), g);
  }
  wait_async();
  __syncthreads();

  const int mi = wid / nw;         // 16-row subtile within block
  const int tn = (wid % nw) << 6;  // 64-col subtile
  const uint16_t* sArow = sAt + (size_t)(mi * 16 + nl) * K;

  v8f acc[4];
#pragma unroll
  for (int i = 0; i < 4; ++i)
#pragma unroll
    for (int j = 0; j < 8; ++j) acc[i][j] = 0.f;

  for (int k0 = 0; k0 < K; k0 += 32) {
    const int kt = k0 >> 5;
    Frag16 af;
#pragma unroll
    for (int vv = 0; vv < 8; ++vv)
      af.u[vv] = *(const uint32_t*)(sArow + k0 + kb_of(vv, half));
#pragma unroll
    for (int i = 0; i < 4; ++i) {
      int n = tn + i * 16 + nl;
      Frag16 bf;
      load_bfrag(bf, Bp, N, kt, n, half);
      acc[i] = __builtin_amdgcn_wmma_f32_16x16x32_bf16(
          false, af.v, false, bf.v, (short)0, acc[i], false, false);
    }
  }

  const int tmg = rowBase + mi * 16;
#pragma unroll
  for (int i = 0; i < 4; ++i) {
    int n = tn + i * 16 + nl;
    float bs = (MODE == 3) ? 0.f : bias[n];
#pragma unroll
    for (int vv = 0; vv < 8; ++vv) {
      int mr = tmg + vv + 8 * half;  // C/D layout: VGPR v -> M=v (+8 hi lanes)
      if (mr < M) {
        float val = acc[i][vv] + bs;
        size_t o = (size_t)mr * N + n;
        if (MODE == 0) {
          outF[o] = val;
        } else if (MODE == 1 || MODE == 3) {
          outB[o] = f2bf(val);
        } else {  // MODE == 2: softplus
          float sp = val > 20.f ? val : log1pf(__expf(val));
          outB[o] = f2bf(sp);
        }
      }
    }
  }
}

// ---------------------------------------------------------------------------
// Node embedding: h = x @ W_atom + b_atom  (K=92, fp32 exact), also bf16 copy.
// ---------------------------------------------------------------------------
__global__ void embed_kernel(const float* __restrict__ x,
                             const float* __restrict__ W,
                             const float* __restrict__ b,
                             float* __restrict__ h, uint16_t* __restrict__ hb) {
  __shared__ float sx[92];
  int n = blockIdx.x;
  int c = threadIdx.x;
  if (c < 92) sx[c] = x[(size_t)n * 92 + c];
  __syncthreads();
  float acc = b[c];
  for (int kk = 0; kk < 92; ++kk) acc += sx[kk] * W[(size_t)kk * 128 + c];
  h[(size_t)n * 128 + c] = acc;
  hb[(size_t)n * 128 + c] = f2bf(acc);
}

// RBF expansion -> bf16 [E,128]
__global__ void rbf_kernel(const float* __restrict__ ea,
                           uint16_t* __restrict__ out, int E) {
  int idx = blockIdx.x * 256 + threadIdx.x;
  if (idx >= E * 128) return;
  int e = idx >> 7;
  int c = idx & 127;
  float center = 8.f * (float)c / 127.f;
  const float gamma = 127.f / 8.f;
  float d = ea[e] - center;
  out[idx] = f2bf(__expf(-gamma * d * d));
}

// ---------------------------------------------------------------------------
// Fused edge kernel. One block = 16 (edge,head) rows.
//   1) async-stage concat(vi,vj,e) (bf16 16x384) into LDS while the VALU
//      computes alpha = qi*[ki|kj|e]*scale into LDS
//   2) LN(384)+sigmoid -> gate (wave32 shuffle reductions)
//   3) WMMA GEMM1: (A_cat @ Wm + bm) * gate -> bf16 back to LDS
//   4) WMMA GEMM2: (gated @ Wmsg + bmsg) -> f32 LDS
//   5) LN(128), atomicAdd into agg[dst, head*128+c]
// ---------------------------------------------------------------------------
__global__ __launch_bounds__(256) void edge_fused_kernel(
    const int* __restrict__ ei, int E, const uint16_t* __restrict__ q,
    const uint16_t* __restrict__ k, const uint16_t* __restrict__ v,
    const uint16_t* __restrict__ eb, const uint32_t* __restrict__ WmP,
    const float* __restrict__ bm, const uint32_t* __restrict__ WmsgP,
    const float* __restrict__ bmsg, const float* __restrict__ lnAg,
    const float* __restrict__ lnAb, const float* __restrict__ lnMg,
    const float* __restrict__ lnMb, float* __restrict__ agg, float scale) {
  __shared__ __align__(16) uint16_t sA[16 * 384];  // A_cat bf16 -> gated bf16
  __shared__ float sG[16 * 384];  // alpha -> gate; later mm f32 (16x128)
  __shared__ int sSrc[16], sDst[16];

  const int tid = threadIdx.x;
  const int wid = tid >> 5;
  const int lane = tid & 31;
  const int nl = lane & 15;
  const int half = lane >> 4;
  const int rowBase = blockIdx.x * 16;  // global row = edge*4 + head

  if (tid < 16) {
    int gr = rowBase + tid;
    int edge = gr >> 2;
    if (edge >= E) edge = E - 1;
    sSrc[tid] = ei[edge];
    sDst[tid] = ei[E + edge];
  }
  __syncthreads();

  // ---- async gather of A_cat = [vi | vj | e] : 768 chunks of 16B ----
  for (int ci = tid; ci < 768; ci += 256) {
    int row = ci / 48;
    int cc = ci - row * 48;
    int seg = cc >> 4;        // 0: vi, 1: vj, 2: e
    int c8 = (cc & 15) << 3;  // element offset within segment
    int head = row & 3;
    int hc = head * 128 + c8;
    int edge = (rowBase + row) >> 2;
    if (edge >= E) edge = E - 1;
    const uint16_t* g;
    if (seg == 0)      g = v + (size_t)sDst[row] * 512 + hc;
    else if (seg == 1) g = v + (size_t)sSrc[row] * 512 + hc;
    else               g = eb + (size_t)edge * 512 + hc;
    async_copy_b128(lds_addr(sA + row * 384 + cc * 8), g);
  }

  // ---- alpha (overlaps with the async copies above) ----
  for (int i = tid; i < 16 * 384; i += 256) {
    int row = i / 384;
    int col = i - row * 384;
    int head = row & 3;
    int seg = col >> 7;
    int c = col & 127;
    int hc = head * 128 + c;
    int dst = sDst[row], src = sSrc[row];
    int edge = (rowBase + row) >> 2;
    if (edge >= E) edge = E - 1;
    float qv = bf2f(q[(size_t)dst * 512 + hc]);
    float kv;
    if (seg == 0)      kv = bf2f(k[(size_t)dst * 512 + hc]);
    else if (seg == 1) kv = bf2f(k[(size_t)src * 512 + hc]);
    else               kv = bf2f(eb[(size_t)edge * 512 + hc]);
    sG[i] = qv * kv * scale;
  }
  wait_async();
  __syncthreads();

  // ---- LN(384) + sigmoid -> gate ----
  for (int rr = 0; rr < 2; ++rr) {
    int row = wid * 2 + rr;
    float s = 0.f, s2 = 0.f;
    for (int j = lane; j < 384; j += 32) {
      float a = sG[row * 384 + j];
      s += a;
      s2 += a * a;
    }
    for (int off = 16; off > 0; off >>= 1) {
      s += __shfl_xor(s, off, 32);
      s2 += __shfl_xor(s2, off, 32);
    }
    float mu = s * (1.f / 384.f);
    float rstd = rsqrtf(s2 * (1.f / 384.f) - mu * mu + 1e-5f);
    for (int j = lane; j < 384; j += 32) {
      float a = (sG[row * 384 + j] - mu) * rstd * lnAg[j] + lnAb[j];
      sG[row * 384 + j] = 1.f / (1.f + __expf(-a));
    }
  }
  __syncthreads();

  // ---- GEMM1: A_cat(16x384) @ Wm(384x384); wave handles 3 n-tiles ----
  v8f acc[3];
#pragma unroll
  for (int i = 0; i < 3; ++i)
#pragma unroll
    for (int j = 0; j < 8; ++j) acc[i][j] = 0.f;

  for (int ks = 0; ks < 12; ++ks) {
    int k0 = ks * 32;
    Frag16 af;
#pragma unroll
    for (int vv = 0; vv < 8; ++vv) {
      int kb = k0 + kb_of(vv, half);
      af.u[vv] = *(const uint32_t*)&sA[nl * 384 + kb];
    }
#pragma unroll
    for (int t = 0; t < 3; ++t) {
      int n = (wid * 3 + t) * 16 + nl;
      Frag16 bf;
      load_bfrag(bf, WmP, 384, ks, n, half);
      acc[t] = __builtin_amdgcn_wmma_f32_16x16x32_bf16(
          false, af.v, false, bf.v, (short)0, acc[t], false, false);
    }
  }
  __syncthreads();  // everyone done reading sA

  // gate + store gated bf16 back into sA
#pragma unroll
  for (int t = 0; t < 3; ++t) {
    int n = (wid * 3 + t) * 16 + nl;
    float bias = bm[n];
#pragma unroll
    for (int vv = 0; vv < 8; ++vv) {
      int mr = vv + 8 * half;
      float g = sG[mr * 384 + n];
      sA[mr * 384 + n] = f2bf((acc[t][vv] + bias) * g);
    }
  }
  __syncthreads();

  // ---- GEMM2: gated(16x384) @ Wmsg(384x128); wave handles 1 n-tile ----
  v8f accm;
#pragma unroll
  for (int j = 0; j < 8; ++j) accm[j] = 0.f;
  for (int ks = 0; ks < 12; ++ks) {
    int k0 = ks * 32;
    Frag16 af, bf;
#pragma unroll
    for (int vv = 0; vv < 8; ++vv) {
      int kb = k0 + kb_of(vv, half);
      af.u[vv] = *(const uint32_t*)&sA[nl * 384 + kb];
    }
    load_bfrag(bf, WmsgP, 128, ks, wid * 16 + nl, half);
    accm = __builtin_amdgcn_wmma_f32_16x16x32_bf16(
        false, af.v, false, bf.v, (short)0, accm, false, false);
  }
  {
    int n = wid * 16 + nl;
    float bias = bmsg[n];
#pragma unroll
    for (int vv = 0; vv < 8; ++vv) {
      int mr = vv + 8 * half;
      sG[mr * 128 + n] = accm[vv] + bias;  // mm f32, reusing gate space
    }
  }
  __syncthreads();

  // ---- LN(128) + scatter-add ----
  for (int rr = 0; rr < 2; ++rr) {
    int row = wid * 2 + rr;
    float s = 0.f, s2 = 0.f;
    for (int j = lane; j < 128; j += 32) {
      float a = sG[row * 128 + j];
      s += a;
      s2 += a * a;
    }
    for (int off = 16; off > 0; off >>= 1) {
      s += __shfl_xor(s, off, 32);
      s2 += __shfl_xor(s2, off, 32);
    }
    float mu = s * (1.f / 128.f);
    float rstd = rsqrtf(s2 * (1.f / 128.f) - mu * mu + 1e-5f);
    int gr = rowBase + row;
    if (gr < E * 4) {
      int dst = sDst[row];
      int head = row & 3;
      for (int j = lane; j < 128; j += 32) {
        float val = (sG[row * 128 + j] - mu) * rstd * lnMg[j] + lnMb[j];
        atomicAdd(&agg[(size_t)dst * 512 + head * 128 + j], val);
      }
    }
  }
}

// ---------------------------------------------------------------------------
// BatchNorm (train mode): per-column mean/rstd over N rows, then apply + SiLU.
// ---------------------------------------------------------------------------
__global__ void bn_stats_kernel(const float* __restrict__ o,
                                float* __restrict__ mu,
                                float* __restrict__ rstd, int N) {
  __shared__ float rs[256], rs2[256];
  int c = blockIdx.x;
  int t = threadIdx.x;
  float s = 0.f, s2 = 0.f;
  for (int n = t; n < N; n += 256) {
    float v = o[(size_t)n * 128 + c];
    s += v;
    s2 += v * v;
  }
  rs[t] = s;
  rs2[t] = s2;
  __syncthreads();
  for (int off = 128; off > 0; off >>= 1) {
    if (t < off) {
      rs[t] += rs[t + off];
      rs2[t] += rs2[t + off];
    }
    __syncthreads();
  }
  if (t == 0) {
    float m = rs[0] / (float)N;
    float var = rs2[0] / (float)N - m * m;
    mu[c] = m;
    rstd[c] = rsqrtf(var + 1e-5f);
  }
}

__global__ void bn_apply_kernel(const float* __restrict__ o,
                                const float* __restrict__ mu,
                                const float* __restrict__ rstd,
                                const float* __restrict__ g,
                                const float* __restrict__ b,
                                float* __restrict__ h,
                                uint16_t* __restrict__ hb, int total) {
  int idx = blockIdx.x * 256 + threadIdx.x;
  if (idx >= total) return;
  int c = idx & 127;
  float v = (o[idx] - mu[c]) * rstd[c] * g[c] + b[c];
  float s = v / (1.f + __expf(-v));  // SiLU
  h[idx] = s;
  hb[idx] = f2bf(s);
}

// ---------------------------------------------------------------------------
// Mean pooling per graph + head MLP with SiLU.
// ---------------------------------------------------------------------------
__global__ void pool_acc_kernel(const float* __restrict__ h,
                                const int* __restrict__ batch,
                                float* __restrict__ pooled,
                                float* __restrict__ counts) {
  int n = blockIdx.x;
  int c = threadIdx.x;
  int g = batch[n];
  atomicAdd(&pooled[(size_t)g * 128 + c], h[(size_t)n * 128 + c]);
  if (c == 0) atomicAdd(&counts[g], 1.f);
}

__global__ void head_kernel(const float* __restrict__ pooled,
                            const float* __restrict__ counts,
                            const float* __restrict__ Wfc,
                            const float* __restrict__ bfc,
                            float* __restrict__ out) {
  __shared__ float p[128];
  int g = blockIdx.x;
  int c = threadIdx.x;
  float cnt = fmaxf(counts[g], 1.f);
  p[c] = pooled[(size_t)g * 128 + c] / cnt;
  __syncthreads();
  float acc = bfc[c];
  for (int kk = 0; kk < 128; ++kk) acc += p[kk] * Wfc[(size_t)kk * 128 + c];
  out[(size_t)g * 128 + c] = acc / (1.f + __expf(-acc));
}

// ---------------------------------------------------------------------------
// Host launch
// ---------------------------------------------------------------------------
extern "C" void kernel_launch(void* const* d_in, const int* in_sizes, int n_in,
                              void* d_out, int out_size, void* d_ws,
                              size_t ws_size, hipStream_t stream) {
  (void)n_in;
  (void)ws_size;
  const float* x = (const float*)d_in[0];
  const float* edge_attr = (const float*)d_in[1];
  const int* edge_index = (const int*)d_in[2];
  const int* batch = (const int*)d_in[3];
  const float* W_atom = (const float*)d_in[4];
  const float* b_atom = (const float*)d_in[5];
  const float* W_rbf1 = (const float*)d_in[6];
  const float* b_rbf1 = (const float*)d_in[7];
  const float* W_rbf2 = (const float*)d_in[8];
  const float* b_rbf2 = (const float*)d_in[9];
  const float* Wq = (const float*)d_in[10];
  const float* bq = (const float*)d_in[11];
  const float* Wk = (const float*)d_in[12];
  const float* bk = (const float*)d_in[13];
  const float* Wv = (const float*)d_in[14];
  const float* bv = (const float*)d_in[15];
  const float* We = (const float*)d_in[16];
  const float* Wc = (const float*)d_in[17];
  const float* bc = (const float*)d_in[18];
  const float* Wm = (const float*)d_in[19];
  const float* bm = (const float*)d_in[20];
  const float* Wmsg = (const float*)d_in[21];
  const float* bmsg = (const float*)d_in[22];
  const float* lnAg = (const float*)d_in[23];
  const float* lnAb = (const float*)d_in[24];
  const float* lnMg = (const float*)d_in[25];
  const float* lnMb = (const float*)d_in[26];
  const float* bng = (const float*)d_in[27];
  const float* bnb = (const float*)d_in[28];
  const float* W_fc = (const float*)d_in[29];
  const float* b_fc = (const float*)d_in[30];

  const int N = in_sizes[0] / 92;
  const int E = in_sizes[1];
  const int G = out_size / 128;
  const int L = 5;
  const float scale = rsqrtf(384.0f);  // 1/sqrt(3*C)

  // ---- workspace carve-out ----
  size_t off = 0;
  char* base = (char*)d_ws;
  auto alloc = [&](size_t bytes) -> void* {
    void* p = base + off;
    off += (bytes + 255) & ~(size_t)255;
    return p;
  };
  float* h = (float*)alloc((size_t)N * 128 * 4);
  uint16_t* hb = (uint16_t*)alloc((size_t)N * 128 * 2);
  uint16_t* rbfb = (uint16_t*)alloc((size_t)E * 128 * 2);
  uint16_t* efb = (uint16_t*)alloc((size_t)E * 128 * 2);
  uint16_t* qb = (uint16_t*)alloc((size_t)N * 512 * 2);
  uint16_t* kb = (uint16_t*)alloc((size_t)N * 512 * 2);
  uint16_t* vb = (uint16_t*)alloc((size_t)N * 512 * 2);
  uint16_t* ebuf = (uint16_t*)alloc((size_t)E * 512 * 2);
  uint16_t* tb = ebuf;  // softplus intermediate aliases ebuf (dead before ebuf)
  float* agg = (float*)alloc((size_t)N * 512 * 4);
  uint16_t* aggb = (uint16_t*)alloc((size_t)N * 512 * 2);
  float* obuf = (float*)alloc((size_t)N * 128 * 4);
  float* mu = (float*)alloc(128 * 4);
  float* rstd = (float*)alloc(128 * 4);
  float* pooled = (float*)alloc((size_t)G * 128 * 4);
  float* counts = (float*)alloc((size_t)G * 4);
  uint32_t* WqP = (uint32_t*)alloc((size_t)L * 64 * 512 * 4);
  uint32_t* WkP = (uint32_t*)alloc((size_t)L * 64 * 512 * 4);
  uint32_t* WvP = (uint32_t*)alloc((size_t)L * 64 * 512 * 4);
  uint32_t* WeP = (uint32_t*)alloc((size_t)L * 64 * 512 * 4);
  uint32_t* WcP = (uint32_t*)alloc((size_t)L * 256 * 128 * 4);
  uint32_t* WmP = (uint32_t*)alloc((size_t)L * 192 * 384 * 4);
  uint32_t* WmsgP = (uint32_t*)alloc((size_t)L * 192 * 128 * 4);
  uint32_t* Wr1P = (uint32_t*)alloc((size_t)64 * 128 * 4);
  uint32_t* Wr2P = (uint32_t*)alloc((size_t)64 * 128 * 4);

  auto pack = [&](const float* W, uint32_t* P, int K, int Nc) {
    int total = (K >> 1) * Nc;
    pack_weight_kernel<<<(total + 255) / 256, 256, 0, stream>>>(W, P, K, Nc);
  };
  auto gemm = [&](int mode, const uint16_t* A, const uint32_t* Bp,
                  const float* bias, float* oF, uint16_t* oB, int M, int Nc,
                  int K) {
    int mw = (Nc == 512) ? 1 : 4;
    int nw = (Nc == 512) ? 8 : 2;
    int blocks = ((M >> 4) + mw - 1) / mw;
    size_t shb = (size_t)mw * 16 * K * 2;
    dim3 g(blocks), blk(256);
    switch (mode) {
      case 0:
        gemm_bf16_kernel<0><<<g, blk, shb, stream>>>(A, Bp, bias, oF, oB, M, Nc,
                                                     K, mw, nw);
        break;
      case 1:
        gemm_bf16_kernel<1><<<g, blk, shb, stream>>>(A, Bp, bias, oF, oB, M, Nc,
                                                     K, mw, nw);
        break;
      case 2:
        gemm_bf16_kernel<2><<<g, blk, shb, stream>>>(A, Bp, bias, oF, oB, M, Nc,
                                                     K, mw, nw);
        break;
      default:
        gemm_bf16_kernel<3><<<g, blk, shb, stream>>>(A, Bp, bias, oF, oB, M, Nc,
                                                     K, mw, nw);
        break;
    }
  };

  // ---- pack all weights to fragment-major bf16 layout ----
  pack(W_rbf1, Wr1P, 128, 128);
  pack(W_rbf2, Wr2P, 128, 128);
  for (int l = 0; l < L; ++l) {
    pack(Wq + (size_t)l * 128 * 512, WqP + (size_t)l * 64 * 512, 128, 512);
    pack(Wk + (size_t)l * 128 * 512, WkP + (size_t)l * 64 * 512, 128, 512);
    pack(Wv + (size_t)l * 128 * 512, WvP + (size_t)l * 64 * 512, 128, 512);
    pack(We + (size_t)l * 128 * 512, WeP + (size_t)l * 64 * 512, 128, 512);
    pack(Wc + (size_t)l * 512 * 128, WcP + (size_t)l * 256 * 128, 512, 128);
    pack(Wm + (size_t)l * 384 * 384, WmP + (size_t)l * 192 * 384, 384, 384);
    pack(Wmsg + (size_t)l * 384 * 128, WmsgP + (size_t)l * 192 * 128, 384, 128);
  }

  // ---- stem: atom embedding (fp32 exact) + RBF MLP (WMMA, softplus) ----
  embed_kernel<<<N, 128, 0, stream>>>(x, W_atom, b_atom, h, hb);
  rbf_kernel<<<(E * 128 + 255) / 256, 256, 0, stream>>>(edge_attr, rbfb, E);
  gemm(2, rbfb, Wr1P, b_rbf1, nullptr, tb, E, 128, 128);  // softplus
  gemm(1, tb, Wr2P, b_rbf2, nullptr, efb, E, 128, 128);   // ef bf16

  // ---- 5 Matformer layers ----
  for (int l = 0; l < L; ++l) {
    gemm(1, hb, WqP + (size_t)l * 64 * 512, bq + (size_t)l * 512, nullptr, qb,
         N, 512, 128);
    gemm(1, hb, WkP + (size_t)l * 64 * 512, bk + (size_t)l * 512, nullptr, kb,
         N, 512, 128);
    gemm(1, hb, WvP + (size_t)l * 64 * 512, bv + (size_t)l * 512, nullptr, vb,
         N, 512, 128);
    gemm(3, efb, WeP + (size_t)l * 64 * 512, nullptr, nullptr, ebuf, E, 512,
         128);

    hipMemsetAsync(agg, 0, (size_t)N * 512 * 4, stream);
    int tiles = (E * 4 + 15) / 16;
    edge_fused_kernel<<<tiles, 256, 0, stream>>>(
        edge_index, E, qb, kb, vb, ebuf, WmP + (size_t)l * 192 * 384,
        bm + (size_t)l * 384, WmsgP + (size_t)l * 192 * 128,
        bmsg + (size_t)l * 128, lnAg + (size_t)l * 384, lnAb + (size_t)l * 384,
        lnMg + (size_t)l * 128, lnMb + (size_t)l * 128, agg, scale);

    f32_to_bf16_kernel<<<(N * 512 + 255) / 256, 256, 0, stream>>>(agg, aggb,
                                                                  N * 512);
    gemm(0, aggb, WcP + (size_t)l * 256 * 128, bc + (size_t)l * 128, obuf,
         nullptr, N, 128, 512);

    bn_stats_kernel<<<128, 256, 0, stream>>>(obuf, mu, rstd, N);
    bn_apply_kernel<<<(N * 128 + 255) / 256, 256, 0, stream>>>(
        obuf, mu, rstd, bng + (size_t)l * 128, bnb + (size_t)l * 128, h, hb,
        N * 128);
  }

  // ---- mean pooling + head ----
  hipMemsetAsync(pooled, 0, (size_t)G * 128 * 4, stream);
  hipMemsetAsync(counts, 0, (size_t)G * 4, stream);
  pool_acc_kernel<<<N, 128, 0, stream>>>(h, batch, pooled, counts);
  head_kernel<<<G, 128, 0, stream>>>(pooled, counts, W_fc, b_fc,
                                     (float*)d_out);
}
